// MambaSequenceModel_64424509440456
// MI455X (gfx1250) — matrix operand order repaired and checked by reading, not compiled, not run on hardware
//
#include <hip/hip_runtime.h>
#include <math.h>

typedef __attribute__((ext_vector_type(16))) _Float16 v16h;
typedef __attribute__((ext_vector_type(8)))  float    v8f;
typedef __attribute__((ext_vector_type(4)))  int      v4i;

#if __has_builtin(__builtin_amdgcn_global_load_async_to_lds_b128) && \
    __has_builtin(__builtin_amdgcn_s_wait_asynccnt)
#define USE_ASYNC 1
typedef __attribute__((address_space(1))) v4i gv4i_t;   // global int4
typedef __attribute__((address_space(3))) v4i lv4i_t;   // LDS int4
#else
#define USE_ASYNC 0
#endif

namespace {
constexpr int kB   = 4;
constexpr int kL   = 2048;
constexpr int kDin = 512;
constexpr int kDm  = 1024;
constexpr int kDi  = 2048;              // d_inner
constexpr int kDs  = 16;                // d_state
constexpr int kDtR = 64;                // dt_rank
constexpr int kXd  = 96;                // dt_rank + 2*d_state
constexpr int kXdP = 128;               // padded x_dbl width (zero cols 96..127)
constexpr int kBL  = kB * kL;           // 8192 rows
constexpr int BK   = 32;                // K slab
}

__device__ __forceinline__ float sigmoidf_(float x) { return 1.f / (1.f + __expf(-x)); }

// ---------------- fp32 -> f16 weight conversion ----------------
__global__ __launch_bounds__(256)
void k_cvt(const float* __restrict__ s, _Float16* __restrict__ d, int n) {
  const int i = blockIdx.x * 256 + threadIdx.x;
  if (i < n) d[i] = (_Float16)s[i];
}

// conversion with zero row padding: rows [rows_src, rows_dst) -> 0
__global__ __launch_bounds__(256)
void k_cvt_pad(const float* __restrict__ s, _Float16* __restrict__ d,
               int rows_src, int rows_dst, int K) {
  const int i = blockIdx.x * 256 + threadIdx.x;
  if (i >= rows_dst * K) return;
  const int r = i / K;
  d[i] = (r < rows_src) ? (_Float16)s[i] : (_Float16)0.f;
}

// ---------------- WMMA GEMM:  C[M,N] = A[M,K](f32) * W[N,K]^T(f16) ----------------
// 256 threads = 8 waves arranged WM(M) x WN(N); block tile (16*WM) x (64*WN).
// Each wave owns a 16x64 strip: 4 accumulators -> 4 back-to-back WMMAs per 32-K slab.
// No bounds logic: grid tiles M and N exactly (N padded where needed).
// A slab (16*WM x 32 f32) double-buffered in LDS via async global->LDS copies.
// act: 0 = none, 1 = softplus.
template<int WM, int WN>
__global__ __launch_bounds__(256)
void k_gemm(const float* __restrict__ A, int lda,
            const _Float16* __restrict__ W,
            float* __restrict__ C, int ldc,
            int K,
            const float* __restrict__ bias, int act)
{
  constexpr int TBM  = 16 * WM;
  constexpr int NCPY = TBM / 32;        // 16B staging copies per thread (1 or 2)
  __shared__ float Asf[2][TBM * BK];

  const int tid  = threadIdx.x;
  const int lane = tid & 31;
  const int wav  = tid >> 5;
  const int mw   = wav / WN;
  const int nw   = wav % WN;
  const int m0   = blockIdx.y * TBM;
  const int n0w  = blockIdx.x * (64 * WN) + nw * 64;

  // 16-bit A fragment as loaded here (contiguous K, same as B): lanes 0-15: M=lane, K 0..15;
  // lanes 16-31: M=lane-16, K 16..31
  const int arow = lane & 15;
  const int aklo = (lane >> 4) * 8;
  // B fragment: lane&15 = N col; lane>=16 -> +16 K; 16 contiguous K halves
  const int bkb  = (lane >> 4) * 16;

  // staging addresses: each thread copies NCPY x 16B of the TBMx32 slab
  const float* sA[NCPY];
  int sdst[NCPY];
#pragma unroll
  for (int i = 0; i < NCPY; ++i) {
    const int e = (tid + i * 256) * 4;
    sdst[i] = e;
    sA[i] = A + (size_t)(m0 + (e >> 5)) * lda + (e & 31);
  }

  const _Float16* wp0 = W + (size_t)(n0w +  0 + arow) * K + bkb;
  const _Float16* wp1 = W + (size_t)(n0w + 16 + arow) * K + bkb;
  const _Float16* wp2 = W + (size_t)(n0w + 32 + arow) * K + bkb;
  const _Float16* wp3 = W + (size_t)(n0w + 48 + arow) * K + bkb;

  v8f acc0 = {}, acc1 = {}, acc2 = {}, acc3 = {};

  auto stage = [&](int buf, int k0) {
#pragma unroll
    for (int i = 0; i < NCPY; ++i) {
#if USE_ASYNC
      __builtin_amdgcn_global_load_async_to_lds_b128(
          (gv4i_t*)(sA[i] + k0), (lv4i_t*)(&Asf[buf][sdst[i]]), 0, 0);
#else
      *(float4*)(&Asf[buf][sdst[i]]) = *(const float4*)(sA[i] + k0);
#endif
    }
  };

  stage(0, 0);
  for (int k0 = 0; k0 < K; k0 += BK) {
    const int cur = (k0 >> 5) & 1;
    const bool more = (k0 + BK < K);
    if (more) stage(cur ^ 1, k0 + BK);
#if USE_ASYNC
    if (more) __builtin_amdgcn_s_wait_asynccnt(1);
    else      __builtin_amdgcn_s_wait_asynccnt(0);
#endif
    __syncthreads();

    const float* as = &Asf[cur][(mw * 16 + arow) * BK];
    v16h af;
#pragma unroll
    for (int j = 0; j < 8; ++j) {
      const int k = 2 * aklo + 2 * j;
      af[2 * j]     = (_Float16)as[k];
      af[2 * j + 1] = (_Float16)as[k + 1];
    }
    const v16h bf0 = *(const v16h*)(wp0 + k0);
    const v16h bf1 = *(const v16h*)(wp1 + k0);
    const v16h bf2 = *(const v16h*)(wp2 + k0);
    const v16h bf3 = *(const v16h*)(wp3 + k0);
    acc0 = __builtin_amdgcn_wmma_f32_16x16x32_f16(false, af, false, bf0, (short)0, acc0, false, false);
    acc1 = __builtin_amdgcn_wmma_f32_16x16x32_f16(false, af, false, bf1, (short)0, acc1, false, false);
    acc2 = __builtin_amdgcn_wmma_f32_16x16x32_f16(false, af, false, bf2, (short)0, acc2, false, false);
    acc3 = __builtin_amdgcn_wmma_f32_16x16x32_f16(false, af, false, bf3, (short)0, acc3, false, false);
    __syncthreads();
  }

  const int mrow0 = m0 + mw * 16 + ((lane >> 4) << 3);    // acc[r] -> row mrow0+r
  const int ncol  = lane & 15;
  auto store_tile = [&](const v8f& a, int nt) {
    const int nn = n0w + nt + ncol;
    const float bv = bias ? bias[nn] : 0.f;
#pragma unroll
    for (int r = 0; r < 8; ++r) {
      float v = a[r] + bv;
      if (act == 1) v = (v > 20.f) ? v : log1pf(__expf(v));   // softplus
      C[(size_t)(mrow0 + r) * ldc + nn] = v;
    }
  };
  store_tile(acc0, 0);
  store_tile(acc1, 16);
  store_tile(acc2, 32);
  store_tile(acc3, 48);
}

// ---------------- depthwise causal conv1d(4) + SiLU ----------------
__global__ __launch_bounds__(256)
void k_conv_silu(const float* __restrict__ u, const float* __restrict__ w,
                 const float* __restrict__ b, float* __restrict__ o)
{
  const size_t idx = (size_t)blockIdx.x * 256 + threadIdx.x;   // over B*L*Di
  const int d = (int)(idx & (kDi - 1));
  const size_t bl = idx >> 11;                                 // /kDi
  const int l = (int)(bl & (kL - 1));
  float acc = b[d];
#pragma unroll
  for (int j = 0; j < 4; ++j) {
    const int t = l - 3 + j;
    if (t >= 0) acc += w[d * 4 + j] * u[(bl - (size_t)(3 - j)) * kDi + d];
  }
  o[idx] = acc * sigmoidf_(acc);
}

// ---------------- selective scan: one thread per (b,d) channel ----------------
__global__ __launch_bounds__(256)
void k_scan(const float* __restrict__ delta, const float* __restrict__ u,
            const float* __restrict__ xdbl, const float* __restrict__ A_log,
            const float* __restrict__ Dp, float* __restrict__ y)
{
  __shared__ float sBC[2 * kDs];
  const int b = blockIdx.x >> 3;                        // kDi/256 = 8 chunks per batch
  const int d = ((blockIdx.x & 7) << 8) + threadIdx.x;
  float a[kDs], h[kDs];
#pragma unroll
  for (int n = 0; n < kDs; ++n) { a[n] = -__expf(A_log[d * kDs + n]); h[n] = 0.f; }
  const float dsk = Dp[d];
  for (int t = 0; t < kL; ++t) {
    const size_t row = (size_t)b * kL + t;
    if (threadIdx.x < 2 * kDs) sBC[threadIdx.x] = xdbl[row * kXdP + kDtR + threadIdx.x];
    __syncthreads();
    const float dl = delta[row * kDi + d];
    const float uv = u[row * kDi + d];
    const float du = dl * uv;
    float acc = 0.f;
#pragma unroll
    for (int n = 0; n < kDs; ++n) {
      h[n] = __expf(dl * a[n]) * h[n] + du * sBC[n];
      acc += h[n] * sBC[kDs + n];
    }
    y[row * kDi + d] = acc + uv * dsk;
    __syncthreads();
  }
}

// ---------------- gating: y *= silu(res) ----------------
__global__ __launch_bounds__(256)
void k_gate(const float* __restrict__ y, const float* __restrict__ r, float* __restrict__ o)
{
  const size_t i = (size_t)blockIdx.x * 256 + threadIdx.x;
  const float rv = r[i];
  o[i] = y[i] * rv * sigmoidf_(rv);
}

// ---------------- in-place LayerNorm over last dim (1024) ----------------
__global__ __launch_bounds__(256)
void k_ln(float* __restrict__ out, const float* __restrict__ g, const float* __restrict__ bt)
{
  __shared__ float s1[256], s2[256];
  float* p = out + (size_t)blockIdx.x * kDm;
  const int tid = threadIdx.x;
  float v[4], s = 0.f, q = 0.f;
#pragma unroll
  for (int i = 0; i < 4; ++i) { v[i] = p[tid + 256 * i]; s += v[i]; q += v[i] * v[i]; }
  s1[tid] = s; s2[tid] = q;
  __syncthreads();
  for (int w = 128; w > 0; w >>= 1) {
    if (tid < w) { s1[tid] += s1[tid + w]; s2[tid] += s2[tid + w]; }
    __syncthreads();
  }
  const float mean = s1[0] * (1.f / kDm);
  const float var  = s2[0] * (1.f / kDm) - mean * mean;
  const float inv  = rsqrtf(var + 1e-5f);
#pragma unroll
  for (int i = 0; i < 4; ++i) {
    const int c = tid + 256 * i;
    p[c] = (v[i] - mean) * inv * g[c] + bt[c];
  }
}

extern "C" void kernel_launch(void* const* d_in, const int* in_sizes, int n_in,
                              void* d_out, int out_size, void* d_ws, size_t ws_size,
                              hipStream_t stream)
{
  (void)in_sizes; (void)n_in; (void)out_size; (void)ws_size;
  const float* inputs = (const float*)d_in[0];
  const float* ipw    = (const float*)d_in[1];   // (1024, 512)
  const float* ipb    = (const float*)d_in[2];
  const float* inpw   = (const float*)d_in[3];   // (4096, 1024)
  const float* convw  = (const float*)d_in[4];   // (2048, 1, 4)
  const float* convb  = (const float*)d_in[5];
  const float* xpw    = (const float*)d_in[6];   // (96, 2048)
  const float* dtw    = (const float*)d_in[7];   // (2048, 64)
  const float* dtb    = (const float*)d_in[8];
  const float* A_log  = (const float*)d_in[9];   // (2048, 16)
  const float* Dp     = (const float*)d_in[10];
  const float* opw    = (const float*)d_in[11];  // (1024, 2048)
  const float* gamma  = (const float*)d_in[12];
  const float* beta   = (const float*)d_in[13];
  float* out = (float*)d_out;

  // ---- workspace carving (256B aligned) ----
  char* ws = (char*)d_ws;
  size_t off = 0;
  auto carve = [&](size_t bytes) -> void* {
    off = (off + 255) & ~(size_t)255;
    void* p = ws + off;
    off += bytes;
    return p;
  };
  _Float16* w_ip = (_Float16*)carve((size_t)kDm  * kDin * 2);
  _Float16* w_in = (_Float16*)carve((size_t)2 * kDi * kDm * 2);
  _Float16* w_xp = (_Float16*)carve((size_t)kXdP * kDi  * 2);   // padded to 128 rows
  _Float16* w_dt = (_Float16*)carve((size_t)kDi  * kDtR * 2);
  _Float16* w_op = (_Float16*)carve((size_t)kDm  * kDi  * 2);
  float* x     = (float*)carve((size_t)kBL * kDm  * 4);
  float* u_raw = (float*)carve((size_t)kBL * kDi  * 4);
  float* res   = (float*)carve((size_t)kBL * kDi  * 4);
  float* u_act = (float*)carve((size_t)kBL * kDi  * 4);
  float* xdbl  = (float*)carve((size_t)kBL * kXdP * 4);         // padded width
  float* delta = (float*)carve((size_t)kBL * kDi  * 4);
  float* yb    = (float*)carve((size_t)kBL * kDi  * 4);

  // ---- 1. convert weights to f16 (stays resident in L2 across GEMMs) ----
  auto cvt = [&](const float* s, _Float16* d, int n) {
    k_cvt<<<(n + 255) / 256, 256, 0, stream>>>(s, d, n);
  };
  cvt(ipw,  w_ip, kDm * kDin);
  cvt(inpw, w_in, 2 * kDi * kDm);
  cvt(dtw,  w_dt, kDi * kDtR);
  cvt(opw,  w_op, kDm * kDi);
  // x_proj: 96 real rows, 32 zero pad rows -> N=128
  k_cvt_pad<<<(kXdP * kDi + 255) / 256, 256, 0, stream>>>(xpw, w_xp, kXd, kXdP, kDi);

  // gemm<WM,WN>: block tile (16*WM) x (64*WN); N and M must divide exactly
  auto gemm24 = [&](const float* A, int lda, const _Float16* W, float* C, int ldc,
                    int M, int N, int K, const float* bias, int act) {
    dim3 grid(N / 256, M / 32);
    k_gemm<2, 4><<<grid, 256, 0, stream>>>(A, lda, W, C, ldc, K, bias, act);
  };
  auto gemm42 = [&](const float* A, int lda, const _Float16* W, float* C, int ldc,
                    int M, int N, int K, const float* bias, int act) {
    dim3 grid(N / 128, M / 64);
    k_gemm<4, 2><<<grid, 256, 0, stream>>>(A, lda, W, C, ldc, K, bias, act);
  };

  // ---- 2. x = inputs @ input_proj_w.T + b ----
  gemm24(inputs, kDin, w_ip, x, kDm, kBL, kDm, kDin, ipb, 0);
  // ---- 3. u_raw = x @ in_proj_w[:Di].T ; res = x @ in_proj_w[Di:].T ----
  gemm24(x, kDm, w_in,                     u_raw, kDi, kBL, kDi, kDm, nullptr, 0);
  gemm24(x, kDm, w_in + (size_t)kDi * kDm, res,   kDi, kBL, kDi, kDm, nullptr, 0);
  // ---- 4. causal depthwise conv + SiLU ----
  k_conv_silu<<<(kBL * kDi) / 256, 256, 0, stream>>>(u_raw, convw, convb, u_act);
  // ---- 5. x_dbl = u @ x_proj_w.T  (N padded to 128, cols 96..127 = 0) ----
  gemm42(u_act, kDi, w_xp, xdbl, kXdP, kBL, kXdP, kDi, nullptr, 0);
  // ---- 6. delta = softplus(x_dbl[:, :64] @ dt_proj_w.T + dt_b) ----
  gemm24(xdbl, kXdP, w_dt, delta, kDi, kBL, kDi, kDtR, dtb, 1);
  // ---- 7. selective scan (y includes + u*D) ----
  k_scan<<<kB * (kDi / 256), 256, 0, stream>>>(delta, u_act, xdbl, A_log, Dp, yb);
  // ---- 8. gate: yg = y * silu(res)  (reuse u_raw) ----
  k_gate<<<(kBL * kDi) / 256, 256, 0, stream>>>(yb, res, u_raw);
  // ---- 9. out = yg @ out_proj_w.T ----
  gemm24(u_raw, kDi, w_op, out, kDm, kBL, kDm, kDi, nullptr, 0);
  // ---- 10. LayerNorm in place ----
  k_ln<<<kBL, 256, 0, stream>>>(out, gamma, beta);
}
